// CustomConv1d_1494648619039
// MI455X (gfx1250) — matrix operand order repaired and checked by reading, not compiled, not run on hardware
//
#include <hip/hip_runtime.h>

// ---------------------------------------------------------------------------
// Conv1d (B=64, Cin=300, L=2048, Cout=512, K=3, pad=1) as implicit GEMM on
// CDNA5 WMMA (bf16 in, f32 accumulate).
// Reduction is k-major with Cin padded to 320: r = k*320 + ci, R' = 960 = 30*32
// => each 32-wide K-step has a CONSTANT kernel tap k, so the im2col B loader
// is a uniform-stride column read (no per-element div/select).
// Weights are prepacked once per launch into d_ws as bf16 A'[512][960]
// (regularizer + zero-pad folded in); the main loop streams the A tile with
// global_load_async_to_lds_b128 (ASYNCcnt) into double-buffered, bank-padded
// LDS. Block tile 128x256, wave tile 64x64 -> 16 WMMAs per K-step per wave.
// ---------------------------------------------------------------------------

typedef __attribute__((ext_vector_type(16))) __bf16 v16bf;
typedef __attribute__((ext_vector_type(8)))  float  v8f;

#define BATCH 64
#define CIN   300
#define CIPAD 320
#define LEN   2048
#define COUT  512
#define KW    3
#define RTOT  (CIN * KW)          // 900
#define RPAD  (CIPAD * KW)        // 960
#define MT    128                 // M tile (Cout)
#define NT    256                 // N tile (L)
#define KT    32                  // WMMA bf16 K
#define KTP   (KT + 8)            // LDS row pitch in bf16 (80B, bank-staggered)
#define RSTEPS (RPAD / KT)        // 30, k constant within each step

union Frag16 { uint4 q[2]; v16bf v; };
union Pack16 { unsigned short s[16]; uint4 q[2]; };
union Pack32 { unsigned short s[32]; uint4 q[4]; };

__device__ __forceinline__ unsigned short f2bf(float f) {
  union { float f; unsigned u; } c; c.f = f;
  unsigned u = c.u;
  u += 0x7FFFu + ((u >> 16) & 1u);            // round-to-nearest-even
  return (unsigned short)(u >> 16);
}

// ---- weight prepack: A'[co][k*320+ci] = w[co,ci,k]*reg[co,k], bf16 ----
__global__ __launch_bounds__(256)
void prepack_weights(const float* __restrict__ w,
                     const float* __restrict__ reg,
                     unsigned short* __restrict__ ap) {
  const int co = blockIdx.x;
  for (int rp = threadIdx.x; rp < RPAD; rp += blockDim.x) {
    const int kk = rp / CIPAD;
    const int ci = rp - kk * CIPAD;
    float v = 0.f;
    if (ci < CIN) v = w[(size_t)co * RTOT + ci * KW + kk] * reg[co * KW + kk];
    ap[(size_t)co * RPAD + rp] = f2bf(v);
  }
}

template <bool PREPACK>
__global__ __launch_bounds__(256)
void conv1d_wmma_bf16(const float* __restrict__ x,
                      const float* __restrict__ w,
                      const float* __restrict__ reg,
                      const float* __restrict__ bias,
                      const unsigned short* __restrict__ aprep,
                      float* __restrict__ out)
{
  // double-buffered tiles, 80B row pitch: A [m][k] bf16, B transposed [n][k]
  __shared__ __align__(16) unsigned short Alds[2][MT * KTP];
  __shared__ __align__(16) unsigned short Blds[2][NT * KTP];

  const int tid  = threadIdx.x;
  const int lane = tid & 31;
  const int wave = tid >> 5;
  const int half = lane >> 4;
  const int l16  = lane & 15;

  // Grid: 4 M-tiles x (64 batches * 8 L-tiles) = 2048 blocks
  const int mt  = blockIdx.x & 3;
  const int nt  = blockIdx.x >> 2;
  const int b   = nt >> 3;
  const int l0  = (nt & 7) * NT;
  const int co0 = mt * MT;

  const int waveM = wave & 1;      // 2 x 64 output rows
  const int waveN = wave >> 1;     // 4 x 64 output cols

  const float* xb = x + (size_t)b * CIN * LEN;

  // loader roles
  const int am = tid >> 1;         // A row (0..127)
  const int ah = tid & 1;          // A 16-wide half of KT
  const int bn = tid;              // B column n (0..255), whole KT column

  float avals[16];
  float bvals[32];

  v8f acc[4][4];
  {
    v8f z = {0.f, 0.f, 0.f, 0.f, 0.f, 0.f, 0.f, 0.f};
    #pragma unroll
    for (int i = 0; i < 4; ++i)
      #pragma unroll
      for (int j = 0; j < 4; ++j) acc[i][j] = z;
  }

  // ---- stage A tile for step rb into buffer buf ----
  auto issueA = [&](int rb, int buf) {
    if (PREPACK) {
      // raw bf16 copy: async global -> LDS, 32B per thread (2 x b128)
      unsigned lds = (unsigned)(size_t)&Alds[buf][am * KTP + ah * 16];
      unsigned long long ga = (unsigned long long)(size_t)
          (aprep + (size_t)(co0 + am) * RPAD + rb * KT + ah * 16);
      asm volatile("global_load_async_to_lds_b128 %0, %1, off"
                   :: "v"(lds), "v"(ga) : "memory");
      asm volatile("global_load_async_to_lds_b128 %0, %1, off offset:16"
                   :: "v"(lds), "v"(ga) : "memory");
    } else {
      const int kk     = rb / (CIPAD / KT);
      const int cibase = rb * KT - kk * CIPAD;
      const int co     = co0 + am;
      const int ci0    = cibase + ah * 16;
      const float rk   = reg[co * KW + kk];
      const float* wp  = w + (size_t)co * RTOT + (size_t)ci0 * KW + kk;
      #pragma unroll
      for (int i = 0; i < 16; ++i)
        avals[i] = (ci0 + i < CIN) ? wp[i * KW] * rk : 0.f;
    }
  };
  auto storeA = [&](int buf) {
    if (!PREPACK) {
      Pack16 p;
      #pragma unroll
      for (int i = 0; i < 16; ++i) p.s[i] = f2bf(avals[i]);
      uint4* dst = (uint4*)&Alds[buf][am * KTP + ah * 16];
      dst[0] = p.q[0]; dst[1] = p.q[1];
    }
  };

  // ---- B tile for step rb: one column per thread, ci stride = LEN ----
  auto loadB = [&](int rb) {
    const int kk     = rb / (CIPAD / KT);
    const int ci0    = rb * KT - kk * CIPAD;
    const int l      = l0 + bn + kk - 1;             // pad = 1
    const float* px  = xb + (size_t)ci0 * LEN + l;
    const bool  lok  = (unsigned)l < (unsigned)LEN;
    if (lok && (ci0 + KT <= CIN)) {
      #pragma unroll
      for (int i = 0; i < KT; ++i) bvals[i] = px[(size_t)i * LEN];
    } else if (lok) {
      #pragma unroll
      for (int i = 0; i < KT; ++i)
        bvals[i] = (ci0 + i < CIN) ? px[(size_t)i * LEN] : 0.f;
    } else {
      #pragma unroll
      for (int i = 0; i < KT; ++i) bvals[i] = 0.f;
    }
  };
  auto storeB = [&](int buf) {
    Pack32 p;
    #pragma unroll
    for (int i = 0; i < KT; ++i) p.s[i] = f2bf(bvals[i]);
    uint4* dst = (uint4*)&Blds[buf][bn * KTP];
    dst[0] = p.q[0]; dst[1] = p.q[1]; dst[2] = p.q[2]; dst[3] = p.q[3];
  };

  // ---- software pipeline: prologue stage 0 ----
  issueA(0, 0);
  loadB(0);
  storeA(0);
  storeB(0);
  if (PREPACK) asm volatile("s_wait_asynccnt 0x0" ::: "memory");
  __syncthreads();

  for (int rb = 0; rb < RSTEPS; ++rb) {
    const int  cur  = rb & 1;
    const int  nxt  = cur ^ 1;
    const bool more = (rb + 1 < RSTEPS);

    if (more) { issueA(rb + 1, nxt); loadB(rb + 1); }

    // fragments per gfx1250 16-bit WMMA layouts (ISA 7.12.2)
    Frag16 afr[4], bfr[4];
    const char* Ab = (const char*)Alds[cur];
    const char* Bb = (const char*)Blds[cur];
    #pragma unroll
    for (int fm = 0; fm < 4; ++fm) {
      const int row  = waveM * 64 + fm * 16 + l16;
      const int base = row * (KTP * 2) + half * 16;   // bytes
      afr[fm].q[0] = *(const uint4*)(Ab + base);
      afr[fm].q[1] = *(const uint4*)(Ab + base + 32);
    }
    #pragma unroll
    for (int fn = 0; fn < 4; ++fn) {
      const int col  = waveN * 64 + fn * 16 + l16;
      const int base = col * (KTP * 2) + half * 16;   // bytes
      bfr[fn].q[0] = *(const uint4*)(Bb + base);
      bfr[fn].q[1] = *(const uint4*)(Bb + base + 32);
    }
    #pragma unroll
    for (int fm = 0; fm < 4; ++fm)
      #pragma unroll
      for (int fn = 0; fn < 4; ++fn)
        acc[fm][fn] = __builtin_amdgcn_wmma_f32_16x16x32_bf16(
            false, afr[fm].v, false, bfr[fn].v,
            (short)0, acc[fm][fn], false, false);

    if (more) { storeA(nxt); storeB(nxt); }
    if (PREPACK) asm volatile("s_wait_asynccnt 0x0" ::: "memory");
    __syncthreads();
  }

  // ---- epilogue: C/D layout -> out[b, co, l] + bias ----
  float* ob = out + (size_t)b * COUT * LEN;
  #pragma unroll
  for (int fm = 0; fm < 4; ++fm) {
    #pragma unroll
    for (int fn = 0; fn < 4; ++fn) {
      const int l = l0 + waveN * 64 + fn * 16 + l16;
      #pragma unroll
      for (int v = 0; v < 8; ++v) {
        const int co = co0 + waveM * 64 + fm * 16 + half * 8 + v;
        ob[(size_t)co * LEN + l] = acc[fm][fn][v] + bias[co];
      }
    }
  }
}

extern "C" void kernel_launch(void* const* d_in, const int* in_sizes, int n_in,
                              void* d_out, int out_size, void* d_ws, size_t ws_size,
                              hipStream_t stream) {
  (void)in_sizes; (void)n_in; (void)out_size;
  const float* x    = (const float*)d_in[0];
  const float* w    = (const float*)d_in[1];
  const float* reg  = (const float*)d_in[2];
  const float* bias = (const float*)d_in[3];
  float* out = (float*)d_out;

  dim3 grid((COUT / MT) * BATCH * (LEN / NT));   // 4 * 64 * 8 = 2048
  dim3 block(256);

  const size_t prepackBytes = (size_t)COUT * RPAD * sizeof(unsigned short); // ~983 KB
  if (d_ws && ws_size >= prepackBytes) {
    unsigned short* ap = (unsigned short*)d_ws;
    prepack_weights<<<dim3(COUT), block, 0, stream>>>(w, reg, ap);
    conv1d_wmma_bf16<true><<<grid, block, 0, stream>>>(x, w, reg, bias, ap, out);
  } else {
    conv1d_wmma_bf16<false><<<grid, block, 0, stream>>>(x, w, reg, bias, nullptr, out);
  }
}
